// EnhancedDynamicFeatureFusion_73400991088877
// MI455X (gfx1250) — matrix operand order repaired and checked by reading, not compiled, not run on hardware
//
#include <hip/hip_runtime.h>
#include <math.h>

// ---------------- problem constants ----------------
#define BB 64
#define CC 1024
#define LL 128
#define HH 8
#define WW 16
#define DD 128
#define NWIN 8
#define MM (BB * LL)        // 8192 rows
#define C4 (4 * CC)         // 4096
#define NQ (CC + HH)        // 1032 (true q_raw width)
#define NQP 1152            // q_raw width padded to multiple of 128
#define ATT_SCALE 0.08838834764831845f  // 1/sqrt(128)

// ---------------- types ----------------
typedef __bf16 bf16_t;
typedef __attribute__((ext_vector_type(16))) bf16_t v16bf;
typedef __attribute__((ext_vector_type(8)))  bf16_t v8bf;
typedef __attribute__((ext_vector_type(8)))  float  v8f;

__device__ __forceinline__ bf16_t f2bf(float f) {
    unsigned u = __builtin_bit_cast(unsigned, f);
    unsigned r = (u + 0x7FFFu + ((u >> 16) & 1u)) >> 16;
    return __builtin_bit_cast(bf16_t, (unsigned short)r);
}

// =====================================================================
// bf16 WMMA GEMM: out = act(A(MxK) @ BT(NxK)^T + bias + resid)
//  - A row-major MxK bf16, BT row-major NxK bf16 (weights pre-transposed)
//  - 128x128 block tile, 256 threads = 8 waves (2x4), each wave 64x32
//    = 4x2 WMMA 16x16x32 tiles; K-step 32; register-prefetch pipeline.
//  - Requires: M%128==0, N%128==0, K%32==0 (callers pad N).
//  - act: 0=none, 1=gelu(exact erf), 2=sigmoid. outF (f32) xor outH (bf16).
// =====================================================================
#define BM 128
#define BN 128
#define BK 32

__global__ __launch_bounds__(256)
void gemm_bf16_kernel(const bf16_t* __restrict__ A, const bf16_t* __restrict__ BT,
                      const float* __restrict__ bias, const float* __restrict__ resid,
                      float* __restrict__ outF, bf16_t* __restrict__ outH,
                      int M, int N, int K, int act)
{
    __shared__ __align__(128) bf16_t As[BM][BK];   // 8 KB
    __shared__ __align__(128) bf16_t Bs[BN][BK];   // 8 KB, Bs[n][k]

    const int tid  = threadIdx.x;
    const int lane = tid & 31;
    const int wv   = tid >> 5;          // 0..7
    const int wm   = wv >> 2;           // 0..1  (64 rows each)
    const int wn   = wv & 3;            // 0..3  (32 cols each)
    const int half = lane >> 4;
    const int l15  = lane & 15;
    const int m0   = blockIdx.y * BM;
    const int n0   = blockIdx.x * BN;

    // staging coords (same pattern for A and B since both are row-major xK)
    const int srow = tid >> 2;          // 0..63 (+64 on pass 1)
    const int scg  = (tid & 3) * 8;     // 0,8,16,24

    v8f acc[4][2];
    #pragma unroll
    for (int i = 0; i < 4; ++i)
        #pragma unroll
        for (int j = 0; j < 2; ++j)
            #pragma unroll
            for (int r = 0; r < 8; ++r) acc[i][j][r] = 0.0f;

    v8bf pa[2], pb[2];
    // prologue: stage k0=0 tile
    #pragma unroll
    for (int p = 0; p < 2; ++p) {
        int r = p * 64 + srow;
        pa[p] = *(const v8bf*)(A  + (size_t)(m0 + r) * K + scg);
        pb[p] = *(const v8bf*)(BT + (size_t)(n0 + r) * K + scg);
    }
    #pragma unroll
    for (int p = 0; p < 2; ++p) {
        int r = p * 64 + srow;
        *(v8bf*)&As[r][scg] = pa[p];
        *(v8bf*)&Bs[r][scg] = pb[p];
    }

    for (int k0 = 0; k0 < K; k0 += BK) {
        __syncthreads();
        const bool more = (k0 + BK) < K;
        if (more) {   // prefetch next tile into registers while computing
            #pragma unroll
            for (int p = 0; p < 2; ++p) {
                int r = p * 64 + srow;
                pa[p] = *(const v8bf*)(A  + (size_t)(m0 + r) * K + (k0 + BK + scg));
                pb[p] = *(const v8bf*)(BT + (size_t)(n0 + r) * K + (k0 + BK + scg));
            }
        }

        // ---- fragments per ISA 7.12.2 layouts ----
        v16bf afr[4], bfr[2];
        #pragma unroll
        for (int ti = 0; ti < 4; ++ti) {
            int m = wm * 64 + ti * 16 + l15;
            // elems 0..7 : K = half*8 + (0..7) ; elems 8..15 : K = 16 + half*8 + (0..7)
            v8bf lo = *(const v8bf*)&As[m][half * 8];
            v8bf hi = *(const v8bf*)&As[m][16 + half * 8];
            afr[ti] = __builtin_shufflevector(lo, hi,
                        0,1,2,3,4,5,6,7,8,9,10,11,12,13,14,15);
        }
        #pragma unroll
        for (int tj = 0; tj < 2; ++tj) {
            int n = wn * 32 + tj * 16 + l15;
            // elems 0..7 : K = half*8 + (0..7) ; elems 8..15 : K = 16 + half*8 + (0..7)
            v8bf blo = *(const v8bf*)&Bs[n][half * 8];
            v8bf bhi = *(const v8bf*)&Bs[n][16 + half * 8];
            bfr[tj] = __builtin_shufflevector(blo, bhi,
                        0,1,2,3,4,5,6,7,8,9,10,11,12,13,14,15);
        }
        #pragma unroll
        for (int ti = 0; ti < 4; ++ti)
            #pragma unroll
            for (int tj = 0; tj < 2; ++tj)
                acc[ti][tj] = __builtin_amdgcn_wmma_f32_16x16x32_bf16(
                    false, afr[ti], false, bfr[tj],
                    (short)0, acc[ti][tj], false, false);

        __syncthreads();
        if (more) {
            #pragma unroll
            for (int p = 0; p < 2; ++p) {
                int r = p * 64 + srow;
                *(v8bf*)&As[r][scg] = pa[p];
                *(v8bf*)&Bs[r][scg] = pb[p];
            }
        }
    }

    // ---- epilogue (D layout: VGPR r -> M = r + 8*half, lane&15 -> N) ----
    #pragma unroll
    for (int ti = 0; ti < 4; ++ti)
        #pragma unroll
        for (int tj = 0; tj < 2; ++tj) {
            int n = n0 + wn * 32 + tj * 16 + l15;
            float bv = bias ? bias[n] : 0.0f;
            #pragma unroll
            for (int r = 0; r < 8; ++r) {
                int m = m0 + wm * 64 + ti * 16 + half * 8 + r;
                float v = acc[ti][tj][r] + bv;
                if (resid) v += resid[(size_t)m * N + n];
                if (act == 1)      v = 0.5f * v * (1.0f + erff(v * 0.70710678118654752f));
                else if (act == 2) v = 1.0f / (1.0f + expf(-v));
                if (outF) outF[(size_t)m * N + n] = v;
                else      outH[(size_t)m * N + n] = f2bf(v);
            }
        }
}

// =====================================================================
// LayerNorm over C=1024, one 256-thread block per row, bf16 output
// =====================================================================
__global__ __launch_bounds__(256)
void layernorm_bf16_kernel(const float* __restrict__ x, const float* __restrict__ g,
                           const float* __restrict__ b, bf16_t* __restrict__ y)
{
    __shared__ float red[256];
    const int row = blockIdx.x, tid = threadIdx.x;
    const float* xr = x + (size_t)row * CC;
    float4 v = *(const float4*)(xr + tid * 4);

    red[tid] = v.x + v.y + v.z + v.w;
    __syncthreads();
    for (int o = 128; o > 0; o >>= 1) { if (tid < o) red[tid] += red[tid + o]; __syncthreads(); }
    float mean = red[0] * (1.0f / CC);
    __syncthreads();

    float d0 = v.x - mean, d1 = v.y - mean, d2 = v.z - mean, d3 = v.w - mean;
    red[tid] = d0 * d0 + d1 * d1 + d2 * d2 + d3 * d3;
    __syncthreads();
    for (int o = 128; o > 0; o >>= 1) { if (tid < o) red[tid] += red[tid + o]; __syncthreads(); }
    float rstd = rsqrtf(red[0] * (1.0f / CC) + 1e-5f);

    int c = tid * 4;
    bf16_t* yr = y + (size_t)row * CC;
    yr[c + 0] = f2bf(d0 * rstd * g[c + 0] + b[c + 0]);
    yr[c + 1] = f2bf(d1 * rstd * g[c + 1] + b[c + 1]);
    yr[c + 2] = f2bf(d2 * rstd * g[c + 2] + b[c + 2]);
    yr[c + 3] = f2bf(d3 * rstd * g[c + 3] + b[c + 3]);
}

// =====================================================================
// Windowed attention + sigmoid gate. One block per (b,h,window).
// q from q_raw (ld=NQP, gate score at col 1024+h), k/v ld=CC. Out (B,L,C) f32.
// =====================================================================
__global__ __launch_bounds__(256)
void attention_kernel(const float* __restrict__ qraw, const float* __restrict__ kbuf,
                      const float* __restrict__ vbuf, float* __restrict__ out)
{
    __shared__ float Qs[WW][DD], Ks[WW][DD], Vs[WW][DD];
    __shared__ float sc[WW][WW];
    __shared__ float gs[WW];

    const int blk = blockIdx.x;
    const int b = blk >> 6;            // / (H*NW)
    const int h = (blk >> 3) & 7;
    const int n = blk & 7;
    const int t0 = n * WW;
    const int tid = threadIdx.x;

    for (int idx = tid; idx < WW * DD; idx += 256) {
        int r = idx >> 7, d = idx & 127;
        size_t tok = (size_t)b * LL + t0 + r;
        Qs[r][d] = qraw[tok * NQP + h * DD + d];
        Ks[r][d] = kbuf[tok * CC + h * DD + d];
        Vs[r][d] = vbuf[tok * CC + h * DD + d];
    }
    if (tid < WW) {
        float gsc = qraw[((size_t)b * LL + t0 + tid) * NQP + CC + h];
        gs[tid] = 1.0f / (1.0f + expf(-gsc));
    }
    __syncthreads();

    const int qi = tid >> 4, ki = tid & 15;
    float s = 0.0f;
    #pragma unroll 8
    for (int d = 0; d < DD; ++d) s += Qs[qi][d] * Ks[ki][d];
    sc[qi][ki] = s * ATT_SCALE;
    __syncthreads();

    float mx = -3.4e38f;
    #pragma unroll
    for (int j = 0; j < WW; ++j) mx = fmaxf(mx, sc[qi][j]);
    float den = 0.0f;
    #pragma unroll
    for (int j = 0; j < WW; ++j) den += expf(sc[qi][j] - mx);
    float p = expf(s * ATT_SCALE - mx) / den;
    __syncthreads();
    sc[qi][ki] = p;
    __syncthreads();

    const int dg = (tid & 15) * 8;
    float o[8];
    #pragma unroll
    for (int j = 0; j < 8; ++j) o[j] = 0.0f;
    #pragma unroll
    for (int j = 0; j < WW; ++j) {
        float pj = sc[qi][j];
        #pragma unroll
        for (int d = 0; d < 8; ++d) o[d] += pj * Vs[j][dg + d];
    }
    float gq = gs[qi];
    size_t base = ((size_t)b * LL + t0 + qi) * CC + h * DD + dg;
    #pragma unroll
    for (int d = 0; d < 8; ++d) out[base + d] = o[d] * gq;
}

// =====================================================================
// Elementwise helpers
// =====================================================================
__global__ void transpose_add_pos_kernel(const float* __restrict__ in,
                                         const float* __restrict__ pos,
                                         float* __restrict__ out)
{
    size_t i = (size_t)blockIdx.x * 256 + threadIdx.x;
    if (i >= (size_t)BB * LL * CC) return;
    int b = (int)(i / (LL * CC));
    int rem = (int)(i % (LL * CC));
    int l = rem / CC, c = rem % CC;
    out[i] = in[((size_t)b * CC + c) * LL + l] + pos[l * CC + c];
}

__global__ void f32_to_bf16_kernel(const float* __restrict__ src,
                                   bf16_t* __restrict__ dst, size_t n)
{
    size_t i = (size_t)blockIdx.x * 256 + threadIdx.x;
    if (i < n) dst[i] = f2bf(src[i]);
}

// W (KxN f32, row-major) -> WT (Npad x K bf16, row-major), zero-pad n>=N
__global__ void transpose_conv_kernel(const float* __restrict__ w,
                                      bf16_t* __restrict__ wt,
                                      int K, int N, int Npad)
{
    size_t i = (size_t)blockIdx.x * 256 + threadIdx.x;
    if (i >= (size_t)Npad * K) return;
    int nn = (int)(i / K), k = (int)(i % K);
    wt[i] = (nn < N) ? f2bf(w[(size_t)k * N + nn]) : __builtin_bit_cast(bf16_t, (unsigned short)0);
}

__global__ void pad_bias_kernel(const float* __restrict__ src, float* __restrict__ dst,
                                int n, int npad)
{
    int i = blockIdx.x * 256 + threadIdx.x;
    if (i < npad) dst[i] = (i < n) ? src[i] : 0.0f;
}

__global__ void concat_bf16_kernel(const float* __restrict__ se, const float* __restrict__ st,
                                   bf16_t* __restrict__ comb)
{
    size_t i = (size_t)blockIdx.x * 256 + threadIdx.x;
    if (i >= (size_t)MM * 2 * CC) return;
    int r = (int)(i / (2 * CC)), c = (int)(i % (2 * CC));
    float v = (c < CC) ? se[(size_t)r * CC + c] : st[(size_t)r * CC + (c - CC)];
    comb[i] = f2bf(v);
}

__global__ void final_gate_kernel(const float* __restrict__ in_seq, const float* __restrict__ in_struct,
                                  const float* __restrict__ se, const float* __restrict__ st,
                                  const float* __restrict__ g,
                                  float* __restrict__ out_seq, float* __restrict__ out_struct)
{
    size_t i = (size_t)blockIdx.x * 256 + threadIdx.x;
    if (i >= (size_t)BB * CC * LL) return;
    int b = (int)(i / (CC * LL));
    int rem = (int)(i % (CC * LL));
    int c = rem / LL, l = rem % LL;
    size_t j = ((size_t)b * LL + l) * CC + c;
    float gv = g[j];
    float sfv = in_seq[i], tfv = in_struct[i];
    out_seq[i]    = sfv + gv * (se[j] - sfv);
    out_struct[i] = tfv + (1.0f - gv) * (st[j] - tfv);
}

// =====================================================================
// Host orchestration
// =====================================================================
struct BlkP {
    const float *nq_g, *nq_b, *nk_g, *nk_b, *nf_g, *nf_b;
    const float *Wq, *bq, *Wk, *bk, *Wv, *bv, *Wo, *bo, *W1, *b1, *W2, *b2;
};
struct WB { bf16_t *wq, *wk, *wv, *wo, *w1, *w2; float* bq_pad; };
struct Scratch {
    bf16_t *xln, *cln;      // cln reused for f_ln; xln reused for attn_bf16
    float *qraw, *kbuf, *vbuf, *attn, *x2;
    bf16_t *hbuf;
};

static inline int ew_grid(size_t n) { return (int)((n + 255) / 256); }

static void run_block(const float* x, const float* ctx, const BlkP& p, const WB& wb,
                      float* se_out, const Scratch& s, hipStream_t stream)
{
    // LayerNorms -> bf16
    layernorm_bf16_kernel<<<MM, 256, 0, stream>>>(x,   p.nq_g, p.nq_b, s.xln);
    layernorm_bf16_kernel<<<MM, 256, 0, stream>>>(ctx, p.nk_g, p.nk_b, s.cln);

    // q_raw = x_ln @ Wq + bq  (padded to N = 1152)
    gemm_bf16_kernel<<<dim3(NQP / BN, MM / BM), 256, 0, stream>>>(
        s.xln, wb.wq, wb.bq_pad, nullptr, s.qraw, nullptr, MM, NQP, CC, 0);
    // k, v = c_ln @ Wk/Wv + b
    gemm_bf16_kernel<<<dim3(CC / BN, MM / BM), 256, 0, stream>>>(
        s.cln, wb.wk, p.bk, nullptr, s.kbuf, nullptr, MM, CC, CC, 0);
    gemm_bf16_kernel<<<dim3(CC / BN, MM / BM), 256, 0, stream>>>(
        s.cln, wb.wv, p.bv, nullptr, s.vbuf, nullptr, MM, CC, CC, 0);

    // windowed attention + gate
    attention_kernel<<<BB * HH * NWIN, 256, 0, stream>>>(s.qraw, s.kbuf, s.vbuf, s.attn);

    // x2 = x + attn @ Wo + bo
    f32_to_bf16_kernel<<<ew_grid((size_t)MM * CC), 256, 0, stream>>>(s.attn, s.xln, (size_t)MM * CC);
    gemm_bf16_kernel<<<dim3(CC / BN, MM / BM), 256, 0, stream>>>(
        s.xln, wb.wo, p.bo, x, s.x2, nullptr, MM, CC, CC, 0);

    // FFN: h = gelu(LN(x2) @ W1 + b1) ; se = x2 + h @ W2 + b2
    layernorm_bf16_kernel<<<MM, 256, 0, stream>>>(s.x2, p.nf_g, p.nf_b, s.cln);
    gemm_bf16_kernel<<<dim3(C4 / BN, MM / BM), 256, 0, stream>>>(
        s.cln, wb.w1, p.b1, nullptr, nullptr, s.hbuf, MM, C4, CC, 1);
    gemm_bf16_kernel<<<dim3(CC / BN, MM / BM), 256, 0, stream>>>(
        s.hbuf, wb.w2, p.b2, s.x2, se_out, nullptr, MM, CC, C4, 0);
}

extern "C" void kernel_launch(void* const* d_in, const int* in_sizes, int n_in,
                              void* d_out, int out_size, void* d_ws, size_t ws_size,
                              hipStream_t stream)
{
    const float* in_seq    = (const float*)d_in[0];
    const float* in_struct = (const float*)d_in[1];
    const float* pos       = (const float*)d_in[2];

    auto get_blk = [&](int o) {
        BlkP p;
        p.nq_g = (const float*)d_in[o + 0];  p.nq_b = (const float*)d_in[o + 1];
        p.nk_g = (const float*)d_in[o + 2];  p.nk_b = (const float*)d_in[o + 3];
        p.nf_g = (const float*)d_in[o + 4];  p.nf_b = (const float*)d_in[o + 5];
        p.Wq = (const float*)d_in[o + 6];    p.bq = (const float*)d_in[o + 7];
        p.Wk = (const float*)d_in[o + 8];    p.bk = (const float*)d_in[o + 9];
        p.Wv = (const float*)d_in[o + 10];   p.bv = (const float*)d_in[o + 11];
        p.Wo = (const float*)d_in[o + 12];   p.bo = (const float*)d_in[o + 13];
        p.W1 = (const float*)d_in[o + 14];   p.b1 = (const float*)d_in[o + 15];
        p.W2 = (const float*)d_in[o + 16];   p.b2 = (const float*)d_in[o + 17];
        return p;
    };
    BlkP pseq = get_blk(3);
    BlkP pstr = get_blk(21);
    const float* Wg = (const float*)d_in[39];
    const float* bg = (const float*)d_in[40];

    // ---- carve workspace ----
    char* wsp = (char*)d_ws;
    size_t off = 0;
    auto take = [&](size_t bytes) -> void* {
        void* p = wsp + off;
        off = (off + bytes + 255) & ~(size_t)255;
        return p;
    };
    const size_t SZ = (size_t)MM * CC;           // 8.39M elements
    float*  seqbuf = (float*)take(SZ * 4);
    float*  strbuf = (float*)take(SZ * 4);
    float*  se     = (float*)take(SZ * 4);
    float*  st     = (float*)take(SZ * 4);
    float*  gbuf   = (float*)take(SZ * 4);
    Scratch s;
    s.qraw = (float*)take((size_t)MM * NQP * 4);
    s.kbuf = (float*)take(SZ * 4);
    s.vbuf = (float*)take(SZ * 4);
    s.attn = (float*)take(SZ * 4);
    s.x2   = (float*)take(SZ * 4);
    s.xln  = (bf16_t*)take(SZ * 2);
    s.cln  = (bf16_t*)take(SZ * 2);
    s.hbuf = (bf16_t*)take((size_t)MM * C4 * 2);
    bf16_t* comb = (bf16_t*)take((size_t)MM * 2 * CC * 2);
    WB wseq, wstr;
    auto take_wb = [&](WB& wb) {
        wb.wq = (bf16_t*)take((size_t)NQP * CC * 2);   // 1152 x 1024 (N x K)
        wb.wk = (bf16_t*)take((size_t)CC * CC * 2);
        wb.wv = (bf16_t*)take((size_t)CC * CC * 2);
        wb.wo = (bf16_t*)take((size_t)CC * CC * 2);
        wb.w1 = (bf16_t*)take((size_t)C4 * CC * 2);    // 4096 x 1024
        wb.w2 = (bf16_t*)take((size_t)CC * C4 * 2);    // 1024 x 4096
        wb.bq_pad = (float*)take((size_t)NQP * 4);
    };
    take_wb(wseq);
    take_wb(wstr);
    bf16_t* wg_t = (bf16_t*)take((size_t)CC * 2 * CC * 2);  // Wg is already N x K
    (void)ws_size; (void)in_sizes; (void)n_in; (void)out_size;

    // ---- weight conversion: all weights -> bf16, transposed to N x K ----
    auto convT = [&](const float* srcp, bf16_t* dstp, int K, int N, int Npad) {
        transpose_conv_kernel<<<ew_grid((size_t)Npad * K), 256, 0, stream>>>(srcp, dstp, K, N, Npad);
    };
    auto conv_blk = [&](const BlkP& p, WB& wb) {
        convT(p.Wq, wb.wq, CC, NQ, NQP);
        convT(p.Wk, wb.wk, CC, CC, CC);
        convT(p.Wv, wb.wv, CC, CC, CC);
        convT(p.Wo, wb.wo, CC, CC, CC);
        convT(p.W1, wb.w1, CC, C4, C4);
        convT(p.W2, wb.w2, C4, CC, CC);
        pad_bias_kernel<<<ew_grid(NQP), 256, 0, stream>>>(p.bq, wb.bq_pad, NQ, NQP);
    };
    conv_blk(pseq, wseq);
    conv_blk(pstr, wstr);
    // comb @ Wg.T : Wg (C x 2C) row-major is exactly the N x K layout we need
    f32_to_bf16_kernel<<<ew_grid((size_t)CC * 2 * CC), 256, 0, stream>>>(Wg, wg_t, (size_t)CC * 2 * CC);

    // ---- prep: seq = sf^T + pos ; struct = tf^T + pos ----
    transpose_add_pos_kernel<<<ew_grid(SZ), 256, 0, stream>>>(in_seq,    pos, seqbuf);
    transpose_add_pos_kernel<<<ew_grid(SZ), 256, 0, stream>>>(in_struct, pos, strbuf);

    // ---- two cross blocks ----
    run_block(seqbuf, strbuf, pseq, wseq, se, s, stream);
    run_block(strbuf, seqbuf, pstr, wstr, st, s, stream);

    // ---- gate: g = sigmoid(concat(se,st) @ Wg^T + bg) ----
    concat_bf16_kernel<<<ew_grid((size_t)MM * 2 * CC), 256, 0, stream>>>(se, st, comb);
    gemm_bf16_kernel<<<dim3(CC / BN, MM / BM), 256, 0, stream>>>(
        comb, wg_t, bg, nullptr, gbuf, nullptr, MM, CC, 2 * CC, 2);

    // ---- final blend + transpose back to (B,C,L) ----
    float* out_seq = (float*)d_out;
    float* out_str = out_seq + (size_t)BB * CC * LL;
    final_gate_kernel<<<ew_grid((size_t)BB * CC * LL), 256, 0, stream>>>(
        in_seq, in_struct, se, st, gbuf, out_seq, out_str);
}
